// BoostedPointPairNet2_73547019976967
// MI455X (gfx1250) — matrix operand order repaired and checked by reading, not compiled, not run on hardware
//
#include <hip/hip_runtime.h>

typedef _Float16 v16h __attribute__((ext_vector_type(16)));
typedef _Float16 v8h  __attribute__((ext_vector_type(8)));
typedef float    v8f  __attribute__((ext_vector_type(8)));

// Padded K-strides (units: halfs). Multiples of 8 (16B align for b128),
// chosen so row->row bank offset avoids LDS bank conflicts.
#define W1_STRIDE 40
#define W2_STRIDE 72
#define W3_STRIDE 136
#define A0_STRIDE 40
#define A1_STRIDE 72
#define A2_STRIDE 136
#define P_STRIDE  264
#define Y1_STRIDE 520
#define Y2_STRIDE 264

// workspace layout (bytes)
#define OFF_W1H 0u
#define OFF_W2H (OFF_W1H + 64u*W1_STRIDE*2u)     // 5120
#define OFF_W3H (OFF_W2H + 128u*W2_STRIDE*2u)    // 23552
#define OFF_V1H (OFF_W3H + 256u*W3_STRIDE*2u)    // 93184
#define OFF_V2H (OFF_V1H + 512u*256u*2u)         // 355328
#define OFF_V3H (OFF_V2H + 256u*512u*2u)         // 617472
#define OFF_P   (OFF_V3H + 48u*256u*2u)          // 642048

__device__ __forceinline__ v8f wmma32f16(v16h a, v16h b, v8f c) {
  return __builtin_amdgcn_wmma_f32_16x16x32_f16(
      /*neg_a=*/false, a, /*neg_b=*/false, b,
      /*c_mod=*/(short)0, c, /*reuse_a=*/false, /*reuse_b=*/false);
}

// A fragment (16x32 f16, MxK): lane L = (h<<4)|m holds, per ISA layout,
// VGPR0-3: K = k0 + h*8 + [0..7], VGPR4-7: K = k0 + 16 + h*8 + [0..7].
__device__ __forceinline__ v16h load_a_frag(const _Float16* base, int stride,
                                            int mrow, int h, int k0) {
  v16h a;
  const _Float16* p = base + mrow * stride + k0 + h * 8;
  *((v8h*)&a)       = *(const v8h*)(p);
  *(((v8h*)&a) + 1) = *(const v8h*)(p + 16);
  return a;
}

// B fragment (32x16 f16, KxN): lane L = (kh<<4)|n holds K = k0 + kh*8 + [0..7], k0 + 16 + kh*8 + [0..7]
// of weight row n (weights stored row-major [out][K], i.e. already B-transposed).
__device__ __forceinline__ v16h load_b_frag(const _Float16* w, int kstride,
                                            int nrow, int kh, int k0) {
  v16h b;
  const _Float16* p = w + nrow * kstride + k0 + kh * 8;
  *((v8h*)&b)       = *(const v8h*)(p);
  *(((v8h*)&b) + 1) = *(const v8h*)(p + 16);
  return b;
}

// float atomic max via ordered int trick (works for LDS and global pointers)
__device__ __forceinline__ void atomic_max_f32(int* addr, float val) {
  int bits = __float_as_int(val);
  if (bits >= 0) atomicMax(addr, bits);
  else           atomicMin((unsigned int*)addr, (unsigned int)bits);
}

// ---------------------------------------------------------------------------
// Kernel 0: convert weights f32 -> f16 (padded layouts), init P to -inf
// ---------------------------------------------------------------------------
__global__ __launch_bounds__(256) void prep_kernel(
    const float* __restrict__ W1, const float* __restrict__ W2,
    const float* __restrict__ W3, const float* __restrict__ V1,
    const float* __restrict__ V2, const float* __restrict__ V3,
    _Float16* __restrict__ W1h, _Float16* __restrict__ W2h,
    _Float16* __restrict__ W3h, _Float16* __restrict__ V1h,
    _Float16* __restrict__ V2h, _Float16* __restrict__ V3h,
    float* __restrict__ P) {
  const int idx = blockIdx.x * blockDim.x + threadIdx.x;
  const int stride = gridDim.x * blockDim.x;
  for (int t = idx; t < 64 * W1_STRIDE; t += stride) {
    int o = t / W1_STRIDE, k = t % W1_STRIDE;
    W1h[t] = (_Float16)((k < 6) ? W1[o * 6 + k] : 0.f);
  }
  for (int t = idx; t < 128 * W2_STRIDE; t += stride) {
    int o = t / W2_STRIDE, k = t % W2_STRIDE;
    W2h[t] = (_Float16)((k < 64) ? W2[o * 64 + k] : 0.f);
  }
  for (int t = idx; t < 256 * W3_STRIDE; t += stride) {
    int o = t / W3_STRIDE, k = t % W3_STRIDE;
    W3h[t] = (_Float16)((k < 128) ? W3[o * 128 + k] : 0.f);
  }
  for (int t = idx; t < 512 * 256; t += stride) V1h[t] = (_Float16)V1[t];
  for (int t = idx; t < 256 * 512; t += stride) V2h[t] = (_Float16)V2[t];
  for (int t = idx; t < 48 * 256; t += stride)
    V3h[t] = (_Float16)((t < 40 * 256) ? V3[t] : 0.f);
  for (int t = idx; t < 16 * 256; t += stride)
    ((int*)P)[t] = 0xFF800000;  // -inf
}

// ---------------------------------------------------------------------------
// Kernel 1: fused pair MLP + max pool.
// grid = 512 blocks: 32 blocks per group; block handles 4 i-rows (iters),
// each of the 8 waves handles 16 consecutive j per i-row (one 16-row WMMA tile).
// ---------------------------------------------------------------------------
__global__ __launch_bounds__(256) void pair_mlp_kernel(
    const float* __restrict__ X,
    const _Float16* __restrict__ gW1, const _Float16* __restrict__ gW2,
    const _Float16* __restrict__ gW3,
    const float* __restrict__ b1, const float* __restrict__ b2,
    float* __restrict__ P) {
  __shared__ alignas(16) _Float16 sW1[64 * W1_STRIDE];
  __shared__ alignas(16) _Float16 sW2[128 * W2_STRIDE];
  __shared__ alignas(16) _Float16 sW3[256 * W3_STRIDE];
  __shared__ alignas(16) _Float16 sAct[8][16 * A0_STRIDE + 16 * A1_STRIDE + 16 * A2_STRIDE];
  __shared__ int sMaxBits[256];

  const int tid = threadIdx.x;

  // stage f16 weights into LDS (16B chunks)
  {
    const uint4* s1 = (const uint4*)gW1;  uint4* d1 = (uint4*)sW1;
    for (int t = tid; t < (64 * W1_STRIDE) / 8; t += 256) d1[t] = s1[t];
    const uint4* s2 = (const uint4*)gW2;  uint4* d2 = (uint4*)sW2;
    for (int t = tid; t < (128 * W2_STRIDE) / 8; t += 256) d2[t] = s2[t];
    const uint4* s3 = (const uint4*)gW3;  uint4* d3 = (uint4*)sW3;
    for (int t = tid; t < (256 * W3_STRIDE) / 8; t += 256) d3[t] = s3[t];
  }
  sMaxBits[tid] = 0xFF800000;
  __syncthreads();

  const int wave = tid >> 5;
  const int lane = tid & 31;
  const int g    = blockIdx.x >> 5;        // 16 groups, 32 blocks each
  const int ib   = (blockIdx.x & 31) * 4;  // base i-row for this block
  const int j0   = wave << 4;              // 16 consecutive j per wave

  _Float16* act0 = sAct[wave];
  _Float16* act1 = act0 + 16 * A0_STRIDE;
  _Float16* act2 = act1 + 16 * A1_STRIDE;

  const int h    = lane >> 4;
  const int mrow = lane & 15;
  const int ncol = lane & 15;

  for (int it = 0; it < 4; ++it) {
    const int i = ib + it;
    // stage input tile: row r = [X[g, j0+r], X[g, i], 0...]
    if (lane < 16) {
      const float* xj = X + (size_t)(g * 128 + j0 + lane) * 3;
      const float* xi = X + (size_t)(g * 128 + i) * 3;
      _Float16* row = act0 + lane * A0_STRIDE;
      row[0] = (_Float16)xj[0]; row[1] = (_Float16)xj[1]; row[2] = (_Float16)xj[2];
      row[3] = (_Float16)xi[0]; row[4] = (_Float16)xi[1]; row[5] = (_Float16)xi[2];
#pragma unroll
      for (int k = 6; k < 32; ++k) row[k] = (_Float16)0.f;
    }

    // ---- layer 1: 6 (padded 32) -> 64 ----
    v16h a0 = load_a_frag(act0, A0_STRIDE, mrow, h, 0);
#pragma unroll
    for (int nt = 0; nt < 4; ++nt) {
      v8f c = {0.f, 0.f, 0.f, 0.f, 0.f, 0.f, 0.f, 0.f};
      c = wmma32f16(a0, load_b_frag(sW1, W1_STRIDE, nt * 16 + ncol, h, 0), c);
      float bias = b1[nt * 16 + ncol];
#pragma unroll
      for (int v = 0; v < 8; ++v)
        act1[(v + 8 * h) * A1_STRIDE + nt * 16 + ncol] =
            (_Float16)fmaxf(c[v] + bias, 0.f);
    }

    // ---- layer 2: 64 -> 128 ----
#pragma unroll
    for (int nt = 0; nt < 8; ++nt) {
      v8f c = {0.f, 0.f, 0.f, 0.f, 0.f, 0.f, 0.f, 0.f};
#pragma unroll
      for (int kt = 0; kt < 2; ++kt)
        c = wmma32f16(load_a_frag(act1, A1_STRIDE, mrow, h, kt * 32),
                      load_b_frag(sW2, W2_STRIDE, nt * 16 + ncol, h, kt * 32), c);
      float bias = b2[nt * 16 + ncol];
#pragma unroll
      for (int v = 0; v < 8; ++v)
        act2[(v + 8 * h) * A2_STRIDE + nt * 16 + ncol] =
            (_Float16)fmaxf(c[v] + bias, 0.f);
    }

    // ---- layer 3: 128 -> 256, fused max-pool (bias b3 deferred to head) ----
#pragma unroll
    for (int nt = 0; nt < 16; ++nt) {
      v8f c = {0.f, 0.f, 0.f, 0.f, 0.f, 0.f, 0.f, 0.f};
#pragma unroll
      for (int kt = 0; kt < 4; ++kt)
        c = wmma32f16(load_a_frag(act2, A2_STRIDE, mrow, h, kt * 32),
                      load_b_frag(sW3, W3_STRIDE, nt * 16 + ncol, h, kt * 32), c);
      float mx = c[0];
#pragma unroll
      for (int v = 1; v < 8; ++v) mx = fmaxf(mx, c[v]);
      atomic_max_f32(&sMaxBits[nt * 16 + ncol], mx);
    }
  }
  __syncthreads();
  // fold block max into per-group global max
  atomic_max_f32((int*)(P + g * 256 + tid), __int_as_float(sMaxBits[tid]));
}

// ---------------------------------------------------------------------------
// Kernel 2: F MLP on pooled P (16 rows = one WMMA M-tile) + boost max -> out
// ---------------------------------------------------------------------------
__global__ __launch_bounds__(256) void boost_head_kernel(
    const float* __restrict__ P, const float* __restrict__ b3,
    const _Float16* __restrict__ V1h, const _Float16* __restrict__ V2h,
    const _Float16* __restrict__ V3h,
    const float* __restrict__ c1, const float* __restrict__ c2,
    const float* __restrict__ c3, float* __restrict__ out) {
  __shared__ alignas(16) _Float16 sP [16 * P_STRIDE];
  __shared__ alignas(16) _Float16 sY1[16 * Y1_STRIDE];
  __shared__ alignas(16) _Float16 sY2[16 * Y2_STRIDE];
  __shared__ float sY3[16 * 48];

  const int tid = threadIdx.x;
  for (int t = tid; t < 16 * 256; t += 256) {
    int r = t >> 8, cix = t & 255;
    sP[r * P_STRIDE + cix] = (_Float16)(P[t] + b3[cix]);  // apply deferred b3
  }
  __syncthreads();

  const int wave = tid >> 5, lane = tid & 31;
  const int h = lane >> 4, mrow = lane & 15, ncol = lane & 15;

  // F1: 256 -> 512 (32 N-tiles, 4 per wave)
  for (int nn = 0; nn < 4; ++nn) {
    const int nt = wave * 4 + nn;
    v8f c = {0.f, 0.f, 0.f, 0.f, 0.f, 0.f, 0.f, 0.f};
#pragma unroll
    for (int kt = 0; kt < 8; ++kt)
      c = wmma32f16(load_a_frag(sP, P_STRIDE, mrow, h, kt * 32),
                    load_b_frag(V1h, 256, nt * 16 + ncol, h, kt * 32), c);
    float bias = c1[nt * 16 + ncol];
#pragma unroll
    for (int v = 0; v < 8; ++v)
      sY1[(v + 8 * h) * Y1_STRIDE + nt * 16 + ncol] =
          (_Float16)fmaxf(c[v] + bias, 0.f);
  }
  __syncthreads();

  // F2: 512 -> 256 (16 N-tiles, 2 per wave)
  for (int nn = 0; nn < 2; ++nn) {
    const int nt = wave * 2 + nn;
    v8f c = {0.f, 0.f, 0.f, 0.f, 0.f, 0.f, 0.f, 0.f};
#pragma unroll
    for (int kt = 0; kt < 16; ++kt)
      c = wmma32f16(load_a_frag(sY1, Y1_STRIDE, mrow, h, kt * 32),
                    load_b_frag(V2h, 512, nt * 16 + ncol, h, kt * 32), c);
    float bias = c2[nt * 16 + ncol];
#pragma unroll
    for (int v = 0; v < 8; ++v)
      sY2[(v + 8 * h) * Y2_STRIDE + nt * 16 + ncol] =
          (_Float16)fmaxf(c[v] + bias, 0.f);
  }
  __syncthreads();

  // F3: 256 -> 48 (40 real), waves 0..2
  if (wave < 3) {
    const int nt = wave;
    v8f c = {0.f, 0.f, 0.f, 0.f, 0.f, 0.f, 0.f, 0.f};
#pragma unroll
    for (int kt = 0; kt < 8; ++kt)
      c = wmma32f16(load_a_frag(sY2, Y2_STRIDE, mrow, h, kt * 32),
                    load_b_frag(V3h, 256, nt * 16 + ncol, h, kt * 32), c);
    const int n = nt * 16 + ncol;
    const float bias = (n < 40) ? c3[n] : 0.f;
#pragma unroll
    for (int v = 0; v < 8; ++v)
      sY3[(v + 8 * h) * 48 + n] = c[v] + bias;
  }
  __syncthreads();

  if (tid < 160) {
    const int b = tid / 40, o = tid % 40;
    float m = sY3[(b * 4 + 0) * 48 + o];
    m = fmaxf(m, sY3[(b * 4 + 1) * 48 + o]);
    m = fmaxf(m, sY3[(b * 4 + 2) * 48 + o]);
    m = fmaxf(m, sY3[(b * 4 + 3) * 48 + o]);
    out[tid] = m;
  }
}

// ---------------------------------------------------------------------------
extern "C" void kernel_launch(void* const* d_in, const int* in_sizes, int n_in,
                              void* d_out, int out_size, void* d_ws, size_t ws_size,
                              hipStream_t stream) {
  (void)in_sizes; (void)n_in; (void)out_size; (void)ws_size;
  const float* X  = (const float*)d_in[0];
  const float* W1 = (const float*)d_in[1];
  const float* b1 = (const float*)d_in[2];
  const float* W2 = (const float*)d_in[3];
  const float* b2 = (const float*)d_in[4];
  const float* W3 = (const float*)d_in[5];
  const float* b3 = (const float*)d_in[6];
  const float* V1 = (const float*)d_in[7];
  const float* c1 = (const float*)d_in[8];
  const float* V2 = (const float*)d_in[9];
  const float* c2 = (const float*)d_in[10];
  const float* V3 = (const float*)d_in[11];
  const float* c3 = (const float*)d_in[12];

  char* ws = (char*)d_ws;
  _Float16* W1h = (_Float16*)(ws + OFF_W1H);
  _Float16* W2h = (_Float16*)(ws + OFF_W2H);
  _Float16* W3h = (_Float16*)(ws + OFF_W3H);
  _Float16* V1h = (_Float16*)(ws + OFF_V1H);
  _Float16* V2h = (_Float16*)(ws + OFF_V2H);
  _Float16* V3h = (_Float16*)(ws + OFF_V3H);
  float*    P   = (float*)   (ws + OFF_P);

  prep_kernel<<<256, 256, 0, stream>>>(W1, W2, W3, V1, V2, V3,
                                       W1h, W2h, W3h, V1h, V2h, V3h, P);
  pair_mlp_kernel<<<512, 256, 0, stream>>>(X, W1h, W2h, W3h, b1, b2, P);
  boost_head_kernel<<<1, 256, 0, stream>>>(P, b3, V1h, V2h, V3h, c1, c2, c3,
                                           (float*)d_out);
}
